// Comb_70016556859799
// MI455X (gfx1250) — matrix operand order repaired and checked by reading, not compiled, not run on hardware
//
#include <hip/hip_runtime.h>

#define BB   4
#define FIN  64
#define FOUT 64
#define HH   128
#define WW   128
#define HO   126
#define WO   126
#define SEGS 8            // ceil(126/16) column segments of 16 pixels
#define RPB  2            // pixel rows per block (two M-tiles share one B fragment)

typedef __attribute__((ext_vector_type(16))) __bf16         v16bf;
typedef __attribute__((ext_vector_type(16))) unsigned short v16u;
typedef __attribute__((ext_vector_type(8)))  unsigned int   v8u;
typedef __attribute__((ext_vector_type(8)))  float          v8f;

__device__ __forceinline__ unsigned short f2bf(float f) {
    unsigned int u = __builtin_bit_cast(unsigned int, f);
    unsigned int r = u + 0x7FFFu + ((u >> 16) & 1u);   // round-to-nearest-even
    return (unsigned short)(r >> 16);
}

// ReLU as integer smax on the IEEE bit pattern: positive floats have int-bits > 0,
// negative floats (incl. -0) have int-bits < 0. Lowers to a single v_max_i32 —
// no canonicalize pair (integer op), no inline-asm scheduling barrier.
__device__ __forceinline__ float relu1(float x) {
    int xi = __builtin_bit_cast(int, x);
    xi = xi > 0 ? xi : 0;
    return __builtin_bit_cast(float, xi);
}

// Full 32-lane B fragments in workspace, one per (i, o_tile):
//   frag[(i*4 + tile)*32 + lane] = 16 bf16 (32 B)
//   column n = lane&15; element j of lane half hf has K = 8*hf + j (j<8), 16 + 8*hf + (j-8) (j>=8):
//       K < 9  -> bf16(Wf[o, i, K])   (conv taps)
//       K == 9 -> bf16(bf[o, i])      (bias row; A supplies 1.0 at K=9)
//       else   -> 0
__global__ void prep_wf_kernel(const float* __restrict__ Wf,
                               const float* __restrict__ bfv,
                               unsigned short* __restrict__ wsB) {
    int idx = blockIdx.x * blockDim.x + threadIdx.x;   // over 64*4*32*16
    if (idx >= FIN * 4 * 32 * 16) return;
    int t    = idx & 15;
    int lane = (idx >> 4) & 31;
    int tile = (idx >> 9) & 3;
    int i    = idx >> 11;
    int o    = tile * 16 + (lane & 15);
    unsigned short v = 0;
    {
        int k = (t < 8) ? (8 * (lane >> 4) + t) : (16 + 8 * (lane >> 4) + (t - 8));
        if (k < 9)       v = f2bf(Wf[(o * FIN + i) * 9 + k]);
        else if (k == 9) v = f2bf(bfv[o * FIN + i]);
    }
    wsB[idx] = v;
}

__global__ __launch_bounds__(128) void dwconv_combine_kernel(
    const float* __restrict__ x, const unsigned short* __restrict__ wsB,
    const float* __restrict__ Wc, const float* __restrict__ bc,
    float* __restrict__ out)
{
    // raw bf16 x tiles: per i, 4 rows x 18 cols (row stride 20, per-i stride 80)
    __shared__ unsigned short xs[FIN * 80];          // 10 KB
    // prebuilt per-lane A fragment halves: [i][row-tile][lane], shared by all waves
    __shared__ uint4 fragA[FIN * 2 * 32];            // 64 KB
    __shared__ float outs[FOUT][RPB][17];            // 8.5 KB, o-major store staging

    int blk = blockIdx.x;
    int b   = blk / ((HO / RPB) * SEGS);
    int rem = blk % ((HO / RPB) * SEGS);
    int h   = (rem / SEGS) * RPB;                    // rows h, h+1
    int seg = rem % SEGS;
    int w0  = seg * 16;

    int tid = threadIdx.x;

    // Pass 1: stage x[b, :, h:h+4, w0:w0+18] into LDS as bf16.
    for (int e = tid; e < FIN * 4 * 18; e += 128) {
        int i   = e / 72;
        int r2  = e % 72;
        int row = r2 / 18;
        int col = r2 % 18;
        int gc  = w0 + col; if (gc > WW - 1) gc = WW - 1;   // clamp right halo
        float v = x[((b * FIN + i) * HH + (h + row)) * WW + gc];
        xs[i * 80 + row * 20 + col] = f2bf(v);
    }
    __syncthreads();

    // Pass 2: im2col into A-fragment layout (16-bit A 16x32), 2 row-tiles per i:
    //   lanes 0-15: elements 0..7 = K=0..7 = taps 0..7 of pixel p
    //   lanes 16-31: element 0 = K=8 = tap 8, element 1 = K=9 = 1.0 (bias row)
    //   elements 8..15 (K=16..31) stay zero in registers.
    for (int e = tid; e < FIN * 2 * 32; e += 128) {
        int i  = e >> 6;
        int r  = e & 63;
        int rt = r >> 5;                 // row-tile 0 -> rows 0..2, 1 -> rows 1..3
        int ls = r & 31;
        int pp = ls & 15;
        const unsigned short* xi = &xs[i * 80 + rt * 20];
        uint4 f;
        if (ls < 16) {
            unsigned int u0 = (unsigned int)xi[pp]          | ((unsigned int)xi[pp + 1]      << 16);
            unsigned int u1 = (unsigned int)xi[pp + 2]      | ((unsigned int)xi[20 + pp]     << 16);
            unsigned int u2 = (unsigned int)xi[20 + pp + 1] | ((unsigned int)xi[20 + pp + 2] << 16);
            unsigned int u3 = (unsigned int)xi[40 + pp]     | ((unsigned int)xi[40 + pp + 1] << 16);
            f = make_uint4(u0, u1, u2, u3);
        } else {
            // [tap8, 1.0bf16, 0, 0, 0, 0, 0, 0]
            f = make_uint4((unsigned int)xi[40 + pp + 2] | (0x3F80u << 16), 0u, 0u, 0u);
        }
        fragA[e] = f;
    }
    __syncthreads();

    int lane   = tid & 31;
    int wv     = tid >> 5;
    int o_base = wv * 16;
    int o      = o_base + (lane & 15);   // this lane's output channel column

    const v16u* wsBv = (const v16u*)wsB;
    const float* wcp = Wc + o * FIN;

    v8f acc0 = {};   // row-tile 0 combine accumulator
    v8f acc1 = {};   // row-tile 1 combine accumulator

    for (int i = 0; i < FIN; ++i) {
        uint4 af0 = fragA[i * 64 + lane];            // ds_load_b128 (row-tile 0)
        uint4 af1 = fragA[i * 64 + 32 + lane];       // ds_load_b128 (row-tile 1)
        v16u bu = wsBv[(i * 4 + wv) * 32 + lane];    // shared B, L2/L0-hot

        v8u aw0 = {af0.x, af0.y, af0.z, af0.w, 0u, 0u, 0u, 0u};
        v8u aw1 = {af1.x, af1.y, af1.z, af1.w, 0u, 0u, 0u, 0u};
        v16bf av0 = __builtin_bit_cast(v16bf, aw0);
        v16bf av1 = __builtin_bit_cast(v16bf, aw1);
        v16bf bv  = __builtin_bit_cast(v16bf, bu);
        v8f cz = {};
        // Two independent WMMAs sharing B: fill each other's result-hazard window.
        v8f c0 = __builtin_amdgcn_wmma_f32_16x16x32_bf16(
            false, av0, false, bv, (short)0, cz, false, false);
        v8f c1 = __builtin_amdgcn_wmma_f32_16x16x32_bf16(
            false, av1, false, bv, (short)0, cz, false, false);

        float wc = wcp[i];
#pragma unroll
        for (int r = 0; r < 8; ++r)
            acc0[r] = __builtin_fmaf(wc, relu1(c0[r]), acc0[r]);   // ReLU + 1x1 combine
#pragma unroll
        for (int r = 0; r < 8; ++r)
            acc1[r] = __builtin_fmaf(wc, relu1(c1[r]), acc1[r]);
    }

    // C layout: VGPR r -> pixel M = r (lanes 0-15) or r+8 (lanes 16-31), N = lane&15.
    int mofs = (lane >= 16) ? 8 : 0;
#pragma unroll
    for (int r = 0; r < 8; ++r) {
        outs[o][0][r + mofs] = acc0[r];
        outs[o][1][r + mofs] = acc1[r];
    }
    __syncthreads();

    // Coalesced store: thread t handles o = t/2, 8 consecutive w (32 B run), both rows.
    {
        int oo    = tid >> 1;
        int c0i   = (tid & 1) * 8;
        float bco = bc[oo];
#pragma unroll
        for (int rt = 0; rt < RPB; ++rt) {
            float* gout = out + ((b * FOUT + oo) * HO + (h + rt)) * WO + w0;
#pragma unroll
            for (int j = 0; j < 8; ++j) {
                int wcol = w0 + c0i + j;
                if (wcol < WO)
                    gout[c0i + j] = outs[oo][rt][c0i + j] + bco;
            }
        }
    }
}

extern "C" void kernel_launch(void* const* d_in, const int* in_sizes, int n_in,
                              void* d_out, int out_size, void* d_ws, size_t ws_size,
                              hipStream_t stream) {
    const float* x  = (const float*)d_in[0];
    const float* Wf = (const float*)d_in[1];
    const float* bf = (const float*)d_in[2];
    const float* Wc = (const float*)d_in[3];
    const float* bc = (const float*)d_in[4];
    float* out = (float*)d_out;
    unsigned short* wsB = (unsigned short*)d_ws;   // needs 64*4*32*16*2 = 256 KB

    prep_wf_kernel<<<(FIN * 4 * 32 * 16 + 255) / 256, 256, 0, stream>>>(Wf, bf, wsB);

    dim3 grid(BB * (HO / RPB) * SEGS);   // 4 * 63 * 8 = 2016 blocks
    dwconv_combine_kernel<<<grid, 128, 0, stream>>>(x, wsB, Wc, bc, out);
}
